// LinearAttention_13718125543935
// MI455X (gfx1250) — compile-verified, matrix operand order checked by reading, not run on hardware
//
#include <hip/hip_runtime.h>
#include <hip/hip_bf16.h>

// ---------------------------------------------------------------------------
// LinearAttention block for MI455X (gfx1250, wave32, WMMA).
// All 4 matmul stages use v_wmma_f32_16x16x32_bf16 (bf16 in, f32 accumulate).
// Layouts chosen so every WMMA fragment but out2's A (from f32 ctx) is two 16-byte vector loads and
// transposed outputs are single 16-byte vector stores.
// ---------------------------------------------------------------------------

#define B_   16
#define C_   256
#define N_   4096      // h*w = 64*64
#define H_   8
#define DH_  32
#define OC3  768       // 3*HIDDEN
#define G_   32
#define CPG  8         // channels per group
#define EPSF 1e-6f

typedef __bf16 bf16;
typedef __attribute__((ext_vector_type(16))) __bf16 bf16x16;
typedef __attribute__((ext_vector_type(8)))  __bf16 bf16x8;
typedef __attribute__((ext_vector_type(8)))  float  f32x8;

// K-slot for 16-bit A/B fragments (ISA 7.12.2):
// elements 0..7  -> K = e        + 8*half
// elements 8..15 -> K = 16+(e-8) + 8*half
__device__ __forceinline__ int kslot(int j, int half) {
    return (j < 8) ? (j + 8 * half) : (j + 8 + 8 * half);
}

// ---------------------------------------------------------------------------
// f32 -> bf16 conversion (weights)
// ---------------------------------------------------------------------------
__global__ void cvt_f32_bf16(const float* __restrict__ src, bf16* __restrict__ dst, int n) {
    int i = blockIdx.x * blockDim.x + threadIdx.x;
    if (i < n) dst[i] = (bf16)src[i];
}

// ---------------------------------------------------------------------------
// x[b][c][n] f32  ->  xT[b][n][c] bf16   (tiled LDS transpose)
// ---------------------------------------------------------------------------
__global__ void cvt_transpose_x(const float* __restrict__ X, bf16* __restrict__ XT) {
    __shared__ float tile[32][33];
    int t  = blockIdx.x;              // b * (C/32) * (N/32) = 16*8*128
    int nt = t & 127;
    int ct = (t >> 7) & 7;
    int b  = t >> 10;
    int tx = threadIdx.x & 31;
    int ty = threadIdx.x >> 5;        // 0..7
    const float* src = X + (size_t)b * C_ * N_ + (size_t)(ct * 32) * N_ + nt * 32;
#pragma unroll
    for (int k = 0; k < 4; ++k)
        tile[ty + 8 * k][tx] = src[(size_t)(ty + 8 * k) * N_ + tx];
    __syncthreads();
    bf16* dst = XT + ((size_t)b * N_ + nt * 32) * C_ + ct * 32;
#pragma unroll
    for (int k = 0; k < 4; ++k)
        dst[(size_t)(ty + 8 * k) * C_ + tx] = (bf16)tile[tx][ty + 8 * k];
}

// ---------------------------------------------------------------------------
// qkv = W_qkv[768,256] x x[b]  using xT[b][n][c].
// q rows (o<256) stored transposed into qT[b][h][n][32] (one b128 store/lane);
// k,v rows stored as KV[b][row][n] for the context GEMM.
// ---------------------------------------------------------------------------
__global__ void qkv_gemm(const bf16* __restrict__ W, const bf16* __restrict__ XT,
                         bf16* __restrict__ QT, bf16* __restrict__ KV) {
    const int lane = threadIdx.x & 31;
    const int wave = threadIdx.x >> 5;
    const int t  = blockIdx.x * 8 + wave;       // B_ * 48 * 256 tiles
    const int nt = t & 255;                     // N/16
    const int ot = (t >> 8) % 48;               // 768/16
    const int b  = t / (256 * 48);
    const int half = lane >> 4;
    const int m    = lane & 15;

    const bf16* Wrow = W + (size_t)(ot * 16 + m) * C_;
    const int   ncol = nt * 16 + m;
    const bf16* Brow = XT + ((size_t)b * N_ + ncol) * C_;

    f32x8 acc = {};
#pragma unroll
    for (int kk = 0; kk < C_ / 32; ++kk) {
        const int kbase = kk * 32;
        bf16x16 a, bfr;
        ((bf16x8*)&a)[0]   = *(const bf16x8*)(Wrow + kbase + 8 * half);
        ((bf16x8*)&a)[1]   = *(const bf16x8*)(Wrow + kbase + 16 + 8 * half);
        ((bf16x8*)&bfr)[0] = *(const bf16x8*)(Brow + kbase + 8 * half);
        ((bf16x8*)&bfr)[1] = *(const bf16x8*)(Brow + kbase + 16 + 8 * half);
        acc = __builtin_amdgcn_wmma_f32_16x16x32_bf16(false, a, false, bfr,
                                                      (short)0, acc, false, false);
    }

    if (ot < 16) {                               // q region -> transposed store
        int h     = ot >> 1;
        int dbase = (ot & 1) * 16 + 8 * half;    // C rows M=i+8*half are consecutive d
        bf16x8 st;
#pragma unroll
        for (int i = 0; i < 8; ++i) st[i] = (bf16)acc[i];
        *(bf16x8*)(QT + ((size_t)(b * H_ + h) * N_ + ncol) * DH_ + dbase) = st;
    } else {                                     // k,v region -> row-major store
        int r = ot * 16 - C_;                    // 0..511
#pragma unroll
        for (int i = 0; i < 8; ++i)
            KV[(size_t)b * 512 * N_ + (size_t)(r + i + 8 * half) * N_ + ncol] = (bf16)acc[i];
    }
}

// ---------------------------------------------------------------------------
// q softmax over d=32 (contiguous in qT) -- one thread per (b,h,n)
// ---------------------------------------------------------------------------
__global__ void q_softmax(bf16* __restrict__ QT) {
    size_t i = (size_t)blockIdx.x * blockDim.x + threadIdx.x;   // B_*H_*N_
    bf16* base = QT + i * DH_;
    bf16x8 r[4];
#pragma unroll
    for (int t = 0; t < 4; ++t) r[t] = *(const bf16x8*)(base + 8 * t);
    float v[DH_];
    float mx = -1e30f;
#pragma unroll
    for (int d = 0; d < DH_; ++d) { v[d] = (float)r[d >> 3][d & 7]; mx = fmaxf(mx, v[d]); }
    float s = 0.f;
#pragma unroll
    for (int d = 0; d < DH_; ++d) { v[d] = __expf(v[d] - mx); s += v[d]; }
    float inv = 1.f / s;
#pragma unroll
    for (int d = 0; d < DH_; ++d) r[d >> 3][d & 7] = (bf16)(v[d] * inv);
#pragma unroll
    for (int t = 0; t < 4; ++t) *(bf16x8*)(base + 8 * t) = r[t];
}

// ---------------------------------------------------------------------------
// k softmax over n (4096, contiguous) -- one block per (b,h,d) row of KV
// ---------------------------------------------------------------------------
__global__ void k_softmax(bf16* __restrict__ KV) {
    __shared__ float red[256];
    int row = blockIdx.x;                  // B_*H_*DH_ = 4096
    int b  = row >> 8;
    int hd = row & 255;
    bf16* base = KV + (size_t)b * 512 * N_ + (size_t)hd * N_;
    int tid = threadIdx.x;

    float mx = -1e30f;
    for (int n = tid; n < N_; n += 256) mx = fmaxf(mx, (float)base[n]);
    red[tid] = mx; __syncthreads();
    for (int s = 128; s > 0; s >>= 1) {
        if (tid < s) red[tid] = fmaxf(red[tid], red[tid + s]);
        __syncthreads();
    }
    mx = red[0]; __syncthreads();

    float sum = 0.f;
    for (int n = tid; n < N_; n += 256) sum += __expf((float)base[n] - mx);
    red[tid] = sum; __syncthreads();
    for (int s = 128; s > 0; s >>= 1) {
        if (tid < s) red[tid] += red[tid + s];
        __syncthreads();
    }
    float inv = 1.f / red[0];
    for (int n = tid; n < N_; n += 256) base[n] = (bf16)(__expf((float)base[n] - mx) * inv);
}

// ---------------------------------------------------------------------------
// context[b,h] (32x32) = k'[32,4096] x v^T[4096,32] -- K=4096, 128 WMMA steps
// ---------------------------------------------------------------------------
__global__ void context_wmma(const bf16* __restrict__ KV, float* __restrict__ CTX) {
    const int lane = threadIdx.x & 31;
    const int wave = threadIdx.x >> 5;      // 0..3
    const int bh = blockIdx.x;              // B_*H_
    const int b = bh >> 3, h = bh & 7;
    const int dm = wave >> 1, en = wave & 1;
    const int half = lane >> 4, m = lane & 15;

    const bf16* Kb = KV + (size_t)b * 512 * N_ + (size_t)(h * DH_) * N_;
    const bf16* Vb = KV + (size_t)b * 512 * N_ + (size_t)(C_ + h * DH_) * N_;
    const bf16* Arow = Kb + (size_t)(dm * 16 + m) * N_;   // A: M=d row, K=n
    const bf16* Brow = Vb + (size_t)(en * 16 + m) * N_;   // B: N=e col, K=n

    f32x8 acc = {};
    for (int kk = 0; kk < N_ / 32; ++kk) {
        const int kbase = kk * 32;
        if (kbase + 512 < N_) {
            __builtin_prefetch(Arow + kbase + 512, 0, 0);   // global_prefetch_b8
            __builtin_prefetch(Brow + kbase + 512, 0, 0);
        }
        bf16x16 a, bfr;
        ((bf16x8*)&a)[0]   = *(const bf16x8*)(Arow + kbase + 8 * half);
        ((bf16x8*)&a)[1]   = *(const bf16x8*)(Arow + kbase + 16 + 8 * half);
        ((bf16x8*)&bfr)[0] = *(const bf16x8*)(Brow + kbase + 8 * half);
        ((bf16x8*)&bfr)[1] = *(const bf16x8*)(Brow + kbase + 16 + 8 * half);
        acc = __builtin_amdgcn_wmma_f32_16x16x32_bf16(false, a, false, bfr,
                                                      (short)0, acc, false, false);
    }
    float* ctx = CTX + (size_t)bh * DH_ * DH_;
#pragma unroll
    for (int i = 0; i < 8; ++i) {
        int d = dm * 16 + i + 8 * half;
        int e = en * 16 + m;
        ctx[d * DH_ + e] = acc[i];
    }
}

// ---------------------------------------------------------------------------
// out[e,n] = sum_d ctx[d,e] * q'[d,n]  (K=32, one WMMA per tile)
// B-fragment from qT (contiguous d); output stored transposed to attT[b][n][c]
// ---------------------------------------------------------------------------
__global__ void out2_wmma(const bf16* __restrict__ QT, const float* __restrict__ CTX,
                          bf16* __restrict__ ATT) {
    const int lane = threadIdx.x & 31;
    const int wave = threadIdx.x >> 5;
    int t  = blockIdx.x * 8 + wave;     // B_*H_*2*256 = 65536
    int nt = t & 255;
    int et = (t >> 8) & 1;
    int h  = (t >> 9) & 7;
    int b  = t >> 12;
    const int half = lane >> 4, m = lane & 15;

    const float* ctx = CTX + (size_t)(b * H_ + h) * DH_ * DH_;
    bf16x16 a;
#pragma unroll
    for (int j = 0; j < 16; ++j) {                  // A[M=e, K=d] = ctx[d,e]
        int d = kslot(j, half);
        a[j] = (bf16)ctx[d * DH_ + (et * 16 + m)];
    }
    int ncol = nt * 16 + m;
    const bf16* Qb = QT + ((size_t)(b * H_ + h) * N_ + ncol) * DH_;
    bf16x16 bfr;
    ((bf16x8*)&bfr)[0] = *(const bf16x8*)(Qb + 8 * half);
    ((bf16x8*)&bfr)[1] = *(const bf16x8*)(Qb + 16 + 8 * half);

    f32x8 acc = {};
    acc = __builtin_amdgcn_wmma_f32_16x16x32_bf16(false, a, false, bfr,
                                                  (short)0, acc, false, false);

    // C rows M = et*16 + i + 8*half are consecutive channels -> one b128 store
    bf16x8 st;
#pragma unroll
    for (int i = 0; i < 8; ++i) st[i] = (bf16)acc[i];
    *(bf16x8*)(ATT + ((size_t)b * N_ + ncol) * C_ + h * DH_ + et * 16 + 8 * half) = st;
}

// ---------------------------------------------------------------------------
// y = W_out[256,256] x att[b] + b_out  using attT[b][n][c] -> f32 Y[b][o][n]
// ---------------------------------------------------------------------------
__global__ void proj_gemm(const bf16* __restrict__ Wo, const bf16* __restrict__ ATT,
                          const float* __restrict__ bias, float* __restrict__ Y) {
    const int lane = threadIdx.x & 31;
    const int wave = threadIdx.x >> 5;
    int t  = blockIdx.x * 8 + wave;     // B_*16*256 = 65536
    int nt = t & 255;
    int ot = (t >> 8) & 15;
    int b  = t >> 12;
    const int half = lane >> 4, m = lane & 15;

    const bf16* Wrow = Wo + (size_t)(ot * 16 + m) * C_;
    int ncol = nt * 16 + m;
    const bf16* Brow = ATT + ((size_t)b * N_ + ncol) * C_;

    f32x8 acc = {};
#pragma unroll
    for (int kk = 0; kk < C_ / 32; ++kk) {
        const int kbase = kk * 32;
        bf16x16 a, bfr;
        ((bf16x8*)&a)[0]   = *(const bf16x8*)(Wrow + kbase + 8 * half);
        ((bf16x8*)&a)[1]   = *(const bf16x8*)(Wrow + kbase + 16 + 8 * half);
        ((bf16x8*)&bfr)[0] = *(const bf16x8*)(Brow + kbase + 8 * half);
        ((bf16x8*)&bfr)[1] = *(const bf16x8*)(Brow + kbase + 16 + 8 * half);
        acc = __builtin_amdgcn_wmma_f32_16x16x32_bf16(false, a, false, bfr,
                                                      (short)0, acc, false, false);
    }
    float* y = Y + (size_t)b * C_ * N_;
#pragma unroll
    for (int i = 0; i < 8; ++i) {
        int o = ot * 16 + i + 8 * half;
        y[(size_t)o * N_ + ncol] = acc[i] + bias[o];
    }
}

// ---------------------------------------------------------------------------
// GroupNorm stats: one block per (b, group) -> mean, var over 8*4096 elems
// ---------------------------------------------------------------------------
__global__ void gn_stats(const float* __restrict__ Y, float* __restrict__ ST) {
    __shared__ float s1[256], s2[256];
    int bg = blockIdx.x;                // B_*G_ = 512
    int b = bg >> 5, g = bg & 31;
    const float* base = Y + (size_t)b * C_ * N_ + (size_t)g * CPG * N_;
    int tid = threadIdx.x;
    float sum = 0.f, sq = 0.f;
    for (int i = tid; i < CPG * N_; i += 256) { float v = base[i]; sum += v; sq += v * v; }
    s1[tid] = sum; s2[tid] = sq; __syncthreads();
    for (int s = 128; s > 0; s >>= 1) {
        if (tid < s) { s1[tid] += s1[tid + s]; s2[tid] += s2[tid + s]; }
        __syncthreads();
    }
    if (tid == 0) {
        float inv_n = 1.f / (float)(CPG * N_);
        float mean = s1[0] * inv_n;
        float var  = s2[0] * inv_n - mean * mean;
        ST[bg * 2]     = mean;
        ST[bg * 2 + 1] = var;
    }
}

// ---------------------------------------------------------------------------
// out = (y - mean) * rsqrt(var+eps) * gamma + beta + x
// ---------------------------------------------------------------------------
__global__ void finalize(const float* __restrict__ Y, const float* __restrict__ ST,
                         const float* __restrict__ gamma, const float* __restrict__ beta,
                         const float* __restrict__ X, float* __restrict__ OUT) {
    size_t i = (size_t)blockIdx.x * blockDim.x + threadIdx.x;  // 2^24 elems
    int c = (int)((i >> 12) & (C_ - 1));
    int b = (int)(i >> 20);
    int g = c / CPG;
    float mean = ST[(b * G_ + g) * 2];
    float var  = ST[(b * G_ + g) * 2 + 1];
    float v = (Y[i] - mean) * __frsqrt_rn(var + EPSF);
    OUT[i] = v * gamma[c] + beta[c] + X[i];
}

// ---------------------------------------------------------------------------
extern "C" void kernel_launch(void* const* d_in, const int* in_sizes, int n_in,
                              void* d_out, int out_size, void* d_ws, size_t ws_size,
                              hipStream_t stream) {
    const float* x      = (const float*)d_in[0];
    const float* w_qkv  = (const float*)d_in[1];
    const float* w_out  = (const float*)d_in[2];
    const float* b_out  = (const float*)d_in[3];
    const float* gamma  = (const float*)d_in[4];
    const float* beta   = (const float*)d_in[5];
    float* out = (float*)d_out;

    // ---- workspace layout (bytes) ----
    char* ws = (char*)d_ws;
    bf16*  xT     = (bf16*)(ws);                      // [b][n][c]      32 MB
    bf16*  wqkvbf = (bf16*)(ws + 33554432);           //               384 KB
    bf16*  woutbf = (bf16*)(ws + 33947648);           //               128 KB
    bf16*  qT     = (bf16*)(ws + 34078720);           // [b][h][n][32]  32 MB
    bf16*  kv     = (bf16*)(ws + 67633152);           // [b][512][n]    64 MB
    float* ctx    = (float*)(ws + 134742016);         //               512 KB
    bf16*  attT   = (bf16*)(ws + 135266304);          // [b][n][256]    32 MB
    float* y      = (float*)(ws + 168820736);         // [b][c][n]      64 MB
    float* stats  = (float*)(ws + 235929600);         //                 4 KB

    // 1) conversions / transpose
    cvt_transpose_x<<<16384, 256, 0, stream>>>(x, xT);
    cvt_f32_bf16<<<768, 256, 0, stream>>>(w_qkv, wqkvbf, OC3 * C_);
    cvt_f32_bf16<<<256, 256, 0, stream>>>(w_out, woutbf, C_ * C_);

    // 2) qkv projection GEMM (WMMA)
    qkv_gemm<<<24576, 256, 0, stream>>>(wqkvbf, xT, qT, kv);

    // 3) softmaxes
    q_softmax<<<2048, 256, 0, stream>>>(qT);
    k_softmax<<<4096, 256, 0, stream>>>(kv);

    // 4) context = k' v^T  (WMMA, K=4096)
    context_wmma<<<B_ * H_, 128, 0, stream>>>(kv, ctx);

    // 5) out = ctx^T q'  (WMMA, K=32) -> attT
    out2_wmma<<<8192, 256, 0, stream>>>(qT, ctx, attT);

    // 6) output projection GEMM + bias (WMMA)
    proj_gemm<<<8192, 256, 0, stream>>>(woutbf, attT, b_out, y);

    // 7) GroupNorm + residual
    gn_stats<<<B_ * G_, 256, 0, stream>>>(y, stats);
    finalize<<<65536, 256, 0, stream>>>(y, stats, gamma, beta, x, out);
}
